// Decoder_20942260535778
// MI455X (gfx1250) — matrix operand order repaired and checked by reading, not compiled, not run on hardware
//
#include <hip/hip_runtime.h>
#include <hip/hip_bf16.h>
#include <cstdint>

#define B_ 8
#define L_ 2500
#define D_ 512
#define Y_ 8921

typedef __attribute__((ext_vector_type(16))) __bf16 v16bf;
typedef __attribute__((ext_vector_type(8)))  float  v8f;

union Frag16 { uint4 q[2]; v16bf v; };

// fp32 -> bf16, round-to-nearest-even, returned as raw u16
__device__ __forceinline__ unsigned short f2bf(float f) {
    union { float f; unsigned int u; } v; v.f = f;
    unsigned int r = v.u + 0x7FFFu + ((v.u >> 16) & 1u);
    return (unsigned short)(r >> 16);
}

__device__ __forceinline__ void pack_store16(unsigned short* dst, const unsigned short t[16]) {
    unsigned int w[8];
#pragma unroll
    for (int j = 0; j < 8; ++j)
        w[j] = (unsigned int)t[2 * j] | ((unsigned int)t[2 * j + 1] << 16);
    *(uint4*)(dst + 0) = make_uint4(w[0], w[1], w[2], w[3]);
    *(uint4*)(dst + 8) = make_uint4(w[4], w[5], w[6], w[7]);
}

// ---------------------------------------------------------------------------
// Kernel 1: scores[b,y,l] = sum_d U_w[y,d] * x[b,l,d]   (written into alpha
// region; softmax'd in place afterwards). 128x128 tile per block, K-step 32.
// ---------------------------------------------------------------------------
__global__ __launch_bounds__(256)
void scores_kernel(const float* __restrict__ x, const float* __restrict__ Uw,
                   float* __restrict__ scores) {
    const int b  = blockIdx.z;
    const int y0 = blockIdx.y * 128;
    const int l0 = blockIdx.x * 128;
    const float* xb = x + (size_t)b * L_ * D_;

    __shared__ __align__(16) unsigned short lsA[128 * 32]; // [m][k] bf16
    __shared__ __align__(16) unsigned short lsB[128 * 32]; // [n][k] bf16 (B^T)

    const int tid   = threadIdx.x;
    const int lane  = tid & 31;
    const int wid   = tid >> 5;
    const int mBase = (wid & 1) * 64;   // 2 waves tile M
    const int nBase = (wid >> 1) * 32;  // 4 waves tile N

    v8f acc[4][2];
#pragma unroll
    for (int i = 0; i < 4; ++i)
#pragma unroll
        for (int j = 0; j < 2; ++j)
            acc[i][j] = (v8f){0.f, 0.f, 0.f, 0.f, 0.f, 0.f, 0.f, 0.f};

    const int row = tid >> 1;          // 0..127
    const int cs  = (tid & 1) * 16;    // 0 or 16

    for (int k0 = 0; k0 < D_; k0 += 32) {
        // stage A: lsA[m][k] = bf16(U_w[y0+m][k0+k])
        {
            unsigned short t[16];
            if (y0 + row < Y_) {
                const float* src = Uw + (size_t)(y0 + row) * D_ + k0 + cs;
#pragma unroll
                for (int j = 0; j < 16; j += 4) {
                    float4 f = *(const float4*)(src + j);
                    t[j + 0] = f2bf(f.x); t[j + 1] = f2bf(f.y);
                    t[j + 2] = f2bf(f.z); t[j + 3] = f2bf(f.w);
                }
                if (k0 + 32 < D_) __builtin_prefetch(src + 32, 0, 1);
            } else {
#pragma unroll
                for (int j = 0; j < 16; ++j) t[j] = 0;
            }
            pack_store16(&lsA[row * 32 + cs], t);
        }
        // stage B^T: lsB[n][k] = bf16(x[b][l0+n][k0+k])
        {
            unsigned short t[16];
            if (l0 + row < L_) {
                const float* src = xb + (size_t)(l0 + row) * D_ + k0 + cs;
#pragma unroll
                for (int j = 0; j < 16; j += 4) {
                    float4 f = *(const float4*)(src + j);
                    t[j + 0] = f2bf(f.x); t[j + 1] = f2bf(f.y);
                    t[j + 2] = f2bf(f.z); t[j + 3] = f2bf(f.w);
                }
                if (k0 + 32 < D_) __builtin_prefetch(src + 32, 0, 1);
            } else {
#pragma unroll
                for (int j = 0; j < 16; ++j) t[j] = 0;
            }
            pack_store16(&lsB[row * 32 + cs], t);
        }
        __syncthreads();

        Frag16 fa[4], fbr[2];
#pragma unroll
        for (int mt = 0; mt < 4; ++mt) {
            int r  = mBase + mt * 16 + (lane & 15);
            int kb = (lane >> 4) * 8;
            fa[mt].q[0] = *(const uint4*)&lsA[r * 32 + kb];
            fa[mt].q[1] = *(const uint4*)&lsA[r * 32 + 16 + kb];
        }
#pragma unroll
        for (int nt = 0; nt < 2; ++nt) {
            int r  = nBase + nt * 16 + (lane & 15);
            int kl = (lane >> 4) * 8;
            fbr[nt].q[0] = *(const uint4*)&lsB[r * 32 + kl];
            fbr[nt].q[1] = *(const uint4*)&lsB[r * 32 + 16 + kl];
        }
#pragma unroll
        for (int mt = 0; mt < 4; ++mt)
#pragma unroll
            for (int nt = 0; nt < 2; ++nt)
                acc[mt][nt] = __builtin_amdgcn_wmma_f32_16x16x32_bf16(
                    false, fa[mt].v, false, fbr[nt].v, (short)0, acc[mt][nt],
                    false, false);
        __syncthreads();
    }

    // C/D layout: VGPR r, lanes 0-15: M=r, N=lane; lanes 16-31: M=8+r, N=lane-16
#pragma unroll
    for (int mt = 0; mt < 4; ++mt) {
        const int rb = y0 + mBase + mt * 16 + ((lane >> 4) << 3);
#pragma unroll
        for (int nt = 0; nt < 2; ++nt) {
            const int c = l0 + nBase + nt * 16 + (lane & 15);
            if (c < L_) {
#pragma unroll
                for (int r = 0; r < 8; ++r) {
                    const int rr = rb + r;
                    if (rr < Y_)
                        scores[((size_t)b * Y_ + rr) * L_ + c] = acc[mt][nt][r];
                }
            }
        }
    }
}

// ---------------------------------------------------------------------------
// Kernel 2: in-place row softmax over L for each (b, y) row of alpha.
// ---------------------------------------------------------------------------
__global__ __launch_bounds__(256)
void softmax_kernel(float* __restrict__ alpha) {
    float* p = alpha + (size_t)blockIdx.x * L_;
    const int tid = threadIdx.x;
    __shared__ float red[256];

    float mx = -3.4e38f;
    for (int i = tid; i < L_; i += 256) mx = fmaxf(mx, p[i]);
    red[tid] = mx; __syncthreads();
    for (int s = 128; s > 0; s >>= 1) {
        if (tid < s) red[tid] = fmaxf(red[tid], red[tid + s]);
        __syncthreads();
    }
    mx = red[0]; __syncthreads();

    float sm = 0.f;
    for (int i = tid; i < L_; i += 256) sm += __expf(p[i] - mx);
    red[tid] = sm; __syncthreads();
    for (int s = 128; s > 0; s >>= 1) {
        if (tid < s) red[tid] += red[tid + s];
        __syncthreads();
    }
    const float inv = 1.0f / red[0];

    for (int i = tid; i < L_; i += 256) p[i] = __expf(p[i] - mx) * inv;
}

// ---------------------------------------------------------------------------
// Kernel 3: m[b,y,d] = sum_l alpha[b,y,l] * x[b,l,d].  M=Y, N=D, K=L (pad 2528)
// ---------------------------------------------------------------------------
__global__ __launch_bounds__(256)
void context_kernel(const float* __restrict__ x, const float* __restrict__ alpha,
                    float* __restrict__ mout) {
    const int b  = blockIdx.z;
    const int y0 = blockIdx.y * 128;
    const int n0 = blockIdx.x * 128;  // d-offset
    const float* xb = x + (size_t)b * L_ * D_;
    const float* ab = alpha + (size_t)b * Y_ * L_;

    __shared__ __align__(16) unsigned short lsA[128 * 32]; // [m][k]
    __shared__ __align__(16) unsigned short lsB[128 * 32]; // [n][k] (B^T)

    const int tid   = threadIdx.x;
    const int lane  = tid & 31;
    const int wid   = tid >> 5;
    const int mBase = (wid & 1) * 64;
    const int nBase = (wid >> 1) * 32;

    v8f acc[4][2];
#pragma unroll
    for (int i = 0; i < 4; ++i)
#pragma unroll
        for (int j = 0; j < 2; ++j)
            acc[i][j] = (v8f){0.f, 0.f, 0.f, 0.f, 0.f, 0.f, 0.f, 0.f};

    const int rowA = tid >> 1;        // 0..127 (y)
    const int csA  = (tid & 1) * 16;  // 0/16 (k)
    const int kkB  = tid >> 3;        // 0..31 (k = l)
    const int nsB  = (tid & 7) * 16;  // 0..112 (n = d)

    for (int k0 = 0; k0 < 2528; k0 += 32) {  // ceil(2500/32)*32
        // stage A: lsA[m][k] = bf16(alpha[y0+m][k0+k]), zero-pad K tail
        {
            unsigned short t[16];
            if (y0 + rowA < Y_) {
                const float* src = ab + (size_t)(y0 + rowA) * L_ + k0 + csA;
#pragma unroll
                for (int j = 0; j < 16; ++j)
                    t[j] = (k0 + csA + j < L_) ? f2bf(src[j]) : (unsigned short)0;
                if (k0 + 32 < L_) __builtin_prefetch(src + 32, 0, 1);
            } else {
#pragma unroll
                for (int j = 0; j < 16; ++j) t[j] = 0;
            }
            pack_store16(&lsA[rowA * 32 + csA], t);
        }
        // stage B^T: lsB[n][k] = bf16(x[b][k0+k][n0+n]); read rows of x
        // (contiguous in d) and transpose into LDS.
        {
            unsigned short t[16];
            if (k0 + kkB < L_) {
                const float* src = xb + (size_t)(k0 + kkB) * D_ + n0 + nsB;
#pragma unroll
                for (int j = 0; j < 16; j += 4) {
                    float4 f = *(const float4*)(src + j);
                    t[j + 0] = f2bf(f.x); t[j + 1] = f2bf(f.y);
                    t[j + 2] = f2bf(f.z); t[j + 3] = f2bf(f.w);
                }
                if (k0 + 32 + kkB < L_)
                    __builtin_prefetch(src + (size_t)32 * D_, 0, 1);
            } else {
#pragma unroll
                for (int j = 0; j < 16; ++j) t[j] = 0;
            }
#pragma unroll
            for (int j = 0; j < 16; ++j) lsB[(nsB + j) * 32 + kkB] = t[j];
        }
        __syncthreads();

        Frag16 fa[4], fbr[2];
#pragma unroll
        for (int mt = 0; mt < 4; ++mt) {
            int r  = mBase + mt * 16 + (lane & 15);
            int kb = (lane >> 4) * 8;
            fa[mt].q[0] = *(const uint4*)&lsA[r * 32 + kb];
            fa[mt].q[1] = *(const uint4*)&lsA[r * 32 + 16 + kb];
        }
#pragma unroll
        for (int nt = 0; nt < 2; ++nt) {
            int r  = nBase + nt * 16 + (lane & 15);
            int kl = (lane >> 4) * 8;
            fbr[nt].q[0] = *(const uint4*)&lsB[r * 32 + kl];
            fbr[nt].q[1] = *(const uint4*)&lsB[r * 32 + 16 + kl];
        }
#pragma unroll
        for (int mt = 0; mt < 4; ++mt)
#pragma unroll
            for (int nt = 0; nt < 2; ++nt)
                acc[mt][nt] = __builtin_amdgcn_wmma_f32_16x16x32_bf16(
                    false, fa[mt].v, false, fbr[nt].v, (short)0, acc[mt][nt],
                    false, false);
        __syncthreads();
    }

#pragma unroll
    for (int mt = 0; mt < 4; ++mt) {
        const int rb = y0 + mBase + mt * 16 + ((lane >> 4) << 3);
#pragma unroll
        for (int nt = 0; nt < 2; ++nt) {
            const int c = n0 + nBase + nt * 16 + (lane & 15);  // always < D_
#pragma unroll
            for (int r = 0; r < 8; ++r) {
                const int rr = rb + r;
                if (rr < Y_)
                    mout[((size_t)b * Y_ + rr) * D_ + c] = acc[mt][nt][r];
            }
        }
    }
}

// ---------------------------------------------------------------------------
// Kernel 4: y[b,yy] = dot(final_w[yy], m[b,yy]) + final_b[yy];  BCE mean loss.
// One wave32 per (b, yy) row.
// ---------------------------------------------------------------------------
__global__ __launch_bounds__(256)
void head_kernel(const float* __restrict__ m, const float* __restrict__ fw,
                 const float* __restrict__ fb, const float* __restrict__ tg,
                 float* __restrict__ yout, float* __restrict__ loss) {
    const int lane = threadIdx.x & 31;
    const int wid  = threadIdx.x >> 5;
    const long long gy = (long long)blockIdx.x * 8 + wid;  // b*Y + yy
    __shared__ float part[8];
    float term = 0.f;

    if (gy < (long long)B_ * Y_) {
        const int yy = (int)(gy % Y_);
        const float* mp = m + gy * D_;
        const float* wp = fw + (size_t)yy * D_;
        float s = 0.f;
#pragma unroll 4
        for (int i = lane; i < D_; i += 32) s += mp[i] * wp[i];
#pragma unroll
        for (int off = 16; off > 0; off >>= 1) s += __shfl_xor(s, off, 32);
        if (lane == 0) {
            const float yv = s + fb[yy];
            yout[gy] = yv;
            const float t = tg[gy];
            term = fmaxf(yv, 0.f) - yv * t + log1pf(__expf(-fabsf(yv)));
        }
    }
    if (lane == 0) part[wid] = term;
    __syncthreads();
    if (threadIdx.x == 0) {
        float ssum = 0.f;
#pragma unroll
        for (int i = 0; i < 8; ++i) ssum += part[i];
        atomicAdd(loss, ssum * (1.0f / ((float)B_ * (float)Y_)));
    }
}

extern "C" void kernel_launch(void* const* d_in, const int* in_sizes, int n_in,
                              void* d_out, int out_size, void* d_ws, size_t ws_size,
                              hipStream_t stream) {
    (void)in_sizes; (void)n_in; (void)out_size; (void)d_ws; (void)ws_size;
    const float* x  = (const float*)d_in[0];
    const float* tg = (const float*)d_in[1];
    // d_in[2] = text_inputs (unused by the reference computation)
    const float* Uw = (const float*)d_in[3];
    const float* fw = (const float*)d_in[4];
    const float* fb = (const float*)d_in[5];

    // d_out layout (flat, return order): y [B,Y] | loss [1] | alpha [B,Y,L] | m [B,Y,D]
    float* out   = (float*)d_out;
    float* yout  = out;
    float* loss  = out + (size_t)B_ * Y_;
    float* alpha = loss + 1;
    float* mbuf  = alpha + (size_t)B_ * Y_ * L_;

    dim3 g1((L_ + 127) / 128, (Y_ + 127) / 128, B_);   // 20 x 70 x 8
    scores_kernel<<<g1, 256, 0, stream>>>(x, Uw, alpha);

    softmax_kernel<<<dim3(B_ * Y_), 256, 0, stream>>>(alpha);

    dim3 g2(D_ / 128, (Y_ + 127) / 128, B_);           // 4 x 70 x 8
    context_kernel<<<g2, 256, 0, stream>>>(x, alpha, mbuf);

    hipMemsetAsync(loss, 0, sizeof(float), stream);
    head_kernel<<<dim3((B_ * Y_) / 8), 256, 0, stream>>>(mbuf, fw, fb, tg, yout, loss);
}
